// Decoder_41437844472056
// MI455X (gfx1250) — matrix operand order repaired and checked by reading, not compiled, not run on hardware
//
#include <hip/hip_runtime.h>
#include <hip/hip_bf16.h>

// ---------------------------------------------------------------------------
// 2-layer GRU decoder (B=1024, H=512, T=24, P=256) for MI455X / gfx1250.
//
//  * Batch rows independent across the recurrence -> 32 WGs x 32 rows, all
//    256 steps local, no inter-WG sync.
//  * Weights repacked once to bf16 WMMA B-fragment layout (4.8 MB, resident
//    in 192 MB L2, shared by all WGs); per-gate lane base pointers + 1KB
//    immediate kt offsets -> no per-fragment 64-bit address math.
//  * 512-thread blocks (16 waves, 4 waves/SIMD). f32 master GRU state lives
//    in LDS (sf0/sf1); bf16 mirrors (sh0/sh1) refreshed per step for WMMA A.
//  * Each wave owns two 16-column slices, forced sequential so only one
//    slice's 4 accumulator tiles is live; accumulators zero-init (inline C=0
//    on first WMMA) with biases folded into the gate epilogue.
//  * Anti-LICM: no __restrict__ + volatile-LDS zero offset in every fragment
//    address so weight loads stream from L2 each step (no hoist+spill).
// ---------------------------------------------------------------------------

#define HID   512
#define TSZ   24
#define TPAD  32
#define BATCH 1024
#define MT    2          // two 16-row M-tiles per block => 32 rows

typedef __bf16 bf16_t;
typedef __attribute__((ext_vector_type(16))) __bf16 v16bf;
typedef __attribute__((ext_vector_type(8)))  float  v8f;

union ABFrag { v16bf bf; uint4 u4[2]; };
union PackPair { __bf16 h[2]; unsigned int u; };

// ------------------------- device helpers ---------------------------------

__device__ __forceinline__ v8f wmma_bf16(v16bf a, v16bf b, v8f c) {
    return __builtin_amdgcn_wmma_f32_16x16x32_bf16(
        false, a, false, b, (short)0, c, false, false);
}

__device__ __forceinline__ v8f zero8() {
    v8f v = {};
    return v;
}

__device__ __forceinline__ float sigmoidf_(float x) {
    return 1.0f / (1.0f + __expf(-x));
}
__device__ __forceinline__ float tanhf_(float x) {
    return 1.0f - 2.0f / (1.0f + __expf(2.0f * x));
}

// A fragment (16x32 bf16, M x K) from row-major LDS activations.
__device__ __forceinline__ v16bf load_a(const bf16_t* s, int m0, int ld,
                                        int k0, int lane) {
    ABFrag f;
    const bf16_t* p = s + (m0 + (lane & 15)) * ld + k0 + ((lane >> 4) << 3);
    f.u4[0] = *(const uint4*)p;
    f.u4[1] = *(const uint4*)(p + 16);
    return f.bf;
}

// Same but with ReLU applied to packed bf16 pairs (sign-mask trick).
__device__ __forceinline__ v16bf load_a_relu(const bf16_t* s, int m0, int ld,
                                             int k0, int lane) {
    ABFrag f;
    const bf16_t* p = s + (m0 + (lane & 15)) * ld + k0 + ((lane >> 4) << 3);
    f.u4[0] = *(const uint4*)p;
    f.u4[1] = *(const uint4*)(p + 16);
    unsigned int* w = (unsigned int*)&f;
#pragma unroll
    for (int i = 0; i < 8; ++i) {
        unsigned int v = w[i];
        unsigned int sgn = v & 0x80008000u;
        unsigned int msk = (sgn - (sgn >> 15)) | sgn;
        w[i] = v & ~msk;
    }
    return f.bf;
}

// Per-(gate,slice) lane base pointer into the packed B stream.
// Tile (nt,kt) occupies 1 KB; kt tiles are contiguous -> immediate offsets.
__device__ __forceinline__ const uint4* bbase(const bf16_t* pack, int nt,
                                              int nKt, int lane, int zoff) {
    return (const uint4*)(pack + zoff +
                          ((size_t)(nt * nKt) * 32 + lane) * 16);
}

// B fragment at kt from a lane base pointer (kt*1024B immediate offset).
__device__ __forceinline__ v16bf load_b_at(const uint4* base, int kt) {
    ABFrag f;
    f.u4[0] = base[kt * 64];
    f.u4[1] = base[kt * 64 + 1];
    return f.bf;
}

// Accumulate one GEMM's contribution to the 3 gate tiles of column-slice ns.
template <int NKT>
__device__ __forceinline__ void gemm3(const bf16_t* act, int ld,
                                      const bf16_t* packW, int ns, int lane,
                                      int zoff, v8f* ar, v8f* az, v8f* an) {
    const uint4* br = bbase(packW, 0 * 32 + ns, NKT, lane, zoff);
    const uint4* bz = bbase(packW, 1 * 32 + ns, NKT, lane, zoff);
    const uint4* bn = bbase(packW, 2 * 32 + ns, NKT, lane, zoff);
#pragma unroll
    for (int kt = 0; kt < NKT; ++kt) {
        v16bf A0 = load_a(act, 0, ld, kt * 32 + zoff, lane);
        v16bf A1 = load_a(act, 16, ld, kt * 32 + zoff, lane);
        v16bf Br = load_b_at(br, kt);
        v16bf Bz = load_b_at(bz, kt);
        v16bf Bn = load_b_at(bn, kt);
        ar[0] = wmma_bf16(A0, Br, ar[0]);
        ar[1] = wmma_bf16(A1, Br, ar[1]);
        az[0] = wmma_bf16(A0, Bz, az[0]);
        az[1] = wmma_bf16(A1, Bz, az[1]);
        an[0] = wmma_bf16(A0, Bn, an[0]);
        an[1] = wmma_bf16(A1, Bn, an[1]);
    }
}

// Refresh a bf16 activation mirror from the f32 LDS master copy.
__device__ __forceinline__ void refresh_bf16(bf16_t* dst, const float* src,
                                             int n, int tid, int nthr) {
    for (int i = tid * 4; i < n; i += nthr * 4) {
        float4 v = *(const float4*)(src + i);
        PackPair p0, p1;
        p0.h[0] = (__bf16)v.x; p0.h[1] = (__bf16)v.y;
        p1.h[0] = (__bf16)v.z; p1.h[1] = (__bf16)v.w;
        uint2 q; q.x = p0.u; q.y = p1.u;
        *(uint2*)(dst + i) = q;
    }
}

// ------------------------- weight pack kernel ------------------------------
__global__ void pack_weights(const float* __restrict__ src,
                             bf16_t* __restrict__ dst,
                             int N, int K, int nKt, int totalU32) {
    unsigned int* out = (unsigned int*)dst;
    for (int u = blockIdx.x * blockDim.x + threadIdx.x; u < totalU32;
         u += gridDim.x * blockDim.x) {
        int j2   = u & 7;
        int lane = (u >> 3) & 31;
        int kt   = (u >> 8) % nKt;
        int nt   = (u >> 8) / nKt;
        int k = kt * 32 + ((lane & 16) ? 8 : 0) + ((j2 & 4) ? 8 : 0) + j2 * 2;
        int n = nt * 16 + (lane & 15);
        float v0 = (n < N && k     < K) ? src[n * K + k]     : 0.0f;
        float v1 = (n < N && (k+1) < K) ? src[n * K + k + 1] : 0.0f;
        PackPair p;
        p.h[0] = (__bf16)v0;
        p.h[1] = (__bf16)v1;
        out[u] = p.u;
    }
}

// ------------------------- main GRU kernel ---------------------------------
// Deliberately NO __restrict__: per-step dout stores must be assumed to alias
// the packed weights so LICM cannot hoist weight loads out of the t-loop.

__global__ __launch_bounds__(512) void gru_wmma_kernel(
    const float* hidden,
    const float* bih0, const float* bhh0,
    const float* bih1, const float* bhh1,
    const float* bout,
    const bf16_t* pWih0, const bf16_t* pWhh0,
    const bf16_t* pWih1, const bf16_t* pWhh1,
    const bf16_t* pWout,
    float* dout, const int* predlen) {
    extern __shared__ char smem[];
    float*  sf0  = (float*)smem;             // 32 x 512 f32 master h0
    float*  sf1  = sf0 + 32 * HID;           // 32 x 512 f32 master h1
    bf16_t* sh0  = (bf16_t*)(sf1 + 32 * HID);// 32 x 512 bf16 mirror
    bf16_t* sh1  = sh0 + 32 * HID;           // 32 x 512 bf16 mirror
    bf16_t* sval = sh1 + 32 * HID;           // 32 x 32 bf16 (cols >=24 = 0)
    int*    szero = (int*)(sval + 32 * TPAD);

    const int tid     = threadIdx.x;
    const int lane    = tid & 31;
    const int w       = tid >> 5;                    // wave id: 0..15
    const int rowBase = blockIdx.x * 32;
    const int P       = *predlen;
    const size_t PT   = (size_t)P * TSZ;
    float* hidf = dout + (size_t)BATCH * PT;

    // ---- init LDS ----
    if (tid == 0) *szero = 0;
    for (int i = tid; i < 32 * TPAD; i += blockDim.x) sval[i] = (bf16_t)0.0f;
    for (int i = tid; i < 32 * HID; i += blockDim.x) {
        int m = i / HID, k = i % HID;
        float v0 = hidden[(size_t)(rowBase + m) * HID + k];
        float v1 = hidden[(size_t)(BATCH + rowBase + m) * HID + k];
        sf0[i] = v0; sh0[i] = (bf16_t)v0;
        sf1[i] = v1; sh1[i] = (bf16_t)v1;
    }

    // ---- per-wave columns: slices ns = 2w, 2w+1 ----
    const int c0 = (2 * w + 0) * 16 + (lane & 15);
    const int c1 = (2 * w + 1) * 16 + (lane & 15);

    // ---- per-column biases as scalars (r/z fused gi+gh; n split) ----
    const float b0r_0  = bih0[c0] + bhh0[c0];
    const float b0r_1  = bih0[c1] + bhh0[c1];
    const float b0z_0  = bih0[HID + c0] + bhh0[HID + c0];
    const float b0z_1  = bih0[HID + c1] + bhh0[HID + c1];
    const float b0in_0 = bih0[2 * HID + c0], b0in_1 = bih0[2 * HID + c1];
    const float b0hn_0 = bhh0[2 * HID + c0], b0hn_1 = bhh0[2 * HID + c1];
    const float b1r_0  = bih1[c0] + bhh1[c0];
    const float b1r_1  = bih1[c1] + bhh1[c1];
    const float b1z_0  = bih1[HID + c0] + bhh1[HID + c0];
    const float b1z_1  = bih1[HID + c1] + bhh1[HID + c1];
    const float b1in_0 = bih1[2 * HID + c0], b1in_1 = bih1[2 * HID + c1];
    const float b1hn_0 = bhh1[2 * HID + c0], b1hn_1 = bhh1[2 * HID + c1];
    float bo = 0.0f;
    if (w < 2) {
        int oc = w * 16 + (lane & 15);
        bo = (oc < TSZ) ? bout[oc] : 0.0f;
    }
    const int rb = (lane >> 4) << 3;       // row sub-offset from acc layout

    __syncthreads();

    for (int t = 0; t < P; ++t) {
        const int zoff = *(volatile int*)szero;    // opaque 0: kills LICM

        // ================= cell 0 =================
#pragma clang loop unroll(disable)
        for (int s = 0; s < 2; ++s) {
            const int ns   = 2 * w + s;
            const int col  = s ? c1 : c0;
            const float br_ = s ? b0r_1 : b0r_0;
            const float bz_ = s ? b0z_1 : b0z_0;
            const float bi_ = s ? b0in_1 : b0in_0;
            const float bh_ = s ? b0hn_1 : b0hn_0;
            v8f ar[MT], az[MT], ain[MT], ahn[MT];
            ar[0] = ar[1] = zero8();
            az[0] = az[1] = zero8();
            ain[0] = ain[1] = zero8();
            ahn[0] = ahn[1] = zero8();
            gemm3<1>(sval, TPAD, pWih0, ns, lane, zoff, ar, az, ain);
            gemm3<16>(sh0, HID, pWhh0, ns, lane, zoff, ar, az, ahn);
#pragma unroll
            for (int mt = 0; mt < MT; ++mt)
#pragma unroll
                for (int i = 0; i < 8; ++i) {
                    int row = mt * 16 + rb + i;
                    float hp = sf0[row * HID + col];
                    float r = sigmoidf_(ar[mt][i] + br_);
                    float z = sigmoidf_(az[mt][i] + bz_);
                    float n = tanhf_((ain[mt][i] + bi_) +
                                     r * (ahn[mt][i] + bh_));
                    sf0[row * HID + col] = (1.0f - z) * n + z * hp;
                }
        }
        __syncthreads();                    // all reads of old sh0 done
        refresh_bf16(sh0, sf0, 32 * HID, tid, blockDim.x);
        __syncthreads();

        // ================= cell 1 =================
#pragma clang loop unroll(disable)
        for (int s = 0; s < 2; ++s) {
            const int ns   = 2 * w + s;
            const int col  = s ? c1 : c0;
            const float br_ = s ? b1r_1 : b1r_0;
            const float bz_ = s ? b1z_1 : b1z_0;
            const float bi_ = s ? b1in_1 : b1in_0;
            const float bh_ = s ? b1hn_1 : b1hn_0;
            v8f ar[MT], az[MT], ain[MT], ahn[MT];
            ar[0] = ar[1] = zero8();
            az[0] = az[1] = zero8();
            ain[0] = ain[1] = zero8();
            ahn[0] = ahn[1] = zero8();
            gemm3<16>(sh0, HID, pWih1, ns, lane, zoff, ar, az, ain);
            gemm3<16>(sh1, HID, pWhh1, ns, lane, zoff, ar, az, ahn);
#pragma unroll
            for (int mt = 0; mt < MT; ++mt)
#pragma unroll
                for (int i = 0; i < 8; ++i) {
                    int row = mt * 16 + rb + i;
                    float hp = sf1[row * HID + col];
                    float r = sigmoidf_(ar[mt][i] + br_);
                    float z = sigmoidf_(az[mt][i] + bz_);
                    float n = tanhf_((ain[mt][i] + bi_) +
                                     r * (ahn[mt][i] + bh_));
                    sf1[row * HID + col] = (1.0f - z) * n + z * hp;
                }
        }
        __syncthreads();                    // all reads of old sh1 done
        refresh_bf16(sh1, sf1, 32 * HID, tid, blockDim.x);
        __syncthreads();

        // ============ output head (waves 0-1) ============
        if (w < 2) {
            v8f o0 = zero8(), o1 = zero8();
            const uint4* bob = bbase(pWout, w, 16, lane, zoff);
#pragma unroll
            for (int kt = 0; kt < 16; ++kt) {
                v16bf A0 = load_a_relu(sh1, 0, HID, kt * 32 + zoff, lane);
                v16bf A1 = load_a_relu(sh1, 16, HID, kt * 32 + zoff, lane);
                v16bf Bo = load_b_at(bob, kt);
                o0 = wmma_bf16(A0, Bo, o0);
                o1 = wmma_bf16(A1, Bo, o1);
            }
            int ocol = w * 16 + (lane & 15);
            if (ocol < TSZ) {
#pragma unroll
                for (int i = 0; i < 8; ++i) {
                    float v0 = sigmoidf_(o0[i] + bo);
                    float v1 = sigmoidf_(o1[i] + bo);
                    int r0 = rb + i, r1 = 16 + rb + i;
                    dout[(size_t)(rowBase + r0) * PT + (size_t)t * TSZ + ocol] = v0;
                    dout[(size_t)(rowBase + r1) * PT + (size_t)t * TSZ + ocol] = v1;
                    sval[r0 * TPAD + ocol] = (bf16_t)v0;
                    sval[r1 * TPAD + ocol] = (bf16_t)v1;
                }
            }
        }
        __syncthreads();                    // sval ready for next step
    }

    // ---- final hidden from f32 master state ----
    for (int i = tid; i < 32 * HID; i += blockDim.x) {
        int m = i / HID, k = i % HID;
        hidf[(size_t)(rowBase + m) * HID + k]         = sf0[i];
        hidf[(size_t)(BATCH + rowBase + m) * HID + k] = sf1[i];
    }
}

// ------------------------- host launcher -----------------------------------

extern "C" void kernel_launch(void* const* d_in, const int* in_sizes, int n_in,
                              void* d_out, int out_size, void* d_ws,
                              size_t ws_size, hipStream_t stream) {
    const float* hidden = (const float*)d_in[0];
    const float* Wih0   = (const float*)d_in[1];
    const float* Whh0   = (const float*)d_in[2];
    const float* bih0   = (const float*)d_in[3];
    const float* bhh0   = (const float*)d_in[4];
    const float* Wih1   = (const float*)d_in[5];
    const float* Whh1   = (const float*)d_in[6];
    const float* bih1   = (const float*)d_in[7];
    const float* bhh1   = (const float*)d_in[8];
    const float* Wout   = (const float*)d_in[9];
    const float* bout   = (const float*)d_in[10];
    const int* predlen  = (const int*)d_in[11];

    char* ws = (char*)d_ws;
    bf16_t* pWhh0 = (bf16_t*)(ws + 0);
    bf16_t* pWih1 = (bf16_t*)(ws + 1572864);
    bf16_t* pWhh1 = (bf16_t*)(ws + 3145728);
    bf16_t* pWih0 = (bf16_t*)(ws + 4718592);
    bf16_t* pWout = (bf16_t*)(ws + 4816896);

    pack_weights<<<512, 256, 0, stream>>>(Whh0, pWhh0, 1536, 512, 16, 393216);
    pack_weights<<<512, 256, 0, stream>>>(Wih1, pWih1, 1536, 512, 16, 393216);
    pack_weights<<<512, 256, 0, stream>>>(Whh1, pWhh1, 1536, 512, 16, 393216);
    pack_weights<<<96, 256, 0, stream>>>(Wih0, pWih0, 1536, 24, 1, 24576);
    pack_weights<<<32, 256, 0, stream>>>(Wout, pWout, 24, 512, 16, 8192);

    // 32 WGs x 512 threads; LDS = 128KB f32 state + 64KB bf16 + 2KB + 16B.
    size_t lds = (size_t)(32 * HID * 4 * 2 + 32 * HID * 2 * 2 +
                          32 * TPAD * 2 + 16);
    gru_wmma_kernel<<<dim3(BATCH / 32), dim3(512), lds, stream>>>(
        hidden, bih0, bhh0, bih1, bhh1, bout,
        pWih0, pWhh0, pWih1, pWhh1, pWout,
        (float*)d_out, predlen);
}
